// MultiHeadAttention_52441550684554
// MI455X (gfx1250) — matrix operand order repaired and checked by reading, not compiled, not run on hardware
//
#include <hip/hip_runtime.h>
#include <hip/hip_bf16.h>
#include <stdint.h>

// ---------------------------------------------------------------------------
// MHA forward for MI455X (gfx1250): bf16 WMMA, fp32 accumulate.
// Double-buffered async global->LDS pipeline; DPP row reductions for softmax.
// 1/sqrt(d) folded into the Q projection epilogue.
// B=4, S=2048, D_MODEL=1024, H=16, d=64.
// ---------------------------------------------------------------------------

typedef __bf16 bf16;
typedef __attribute__((ext_vector_type(16))) __bf16 v16bf;
typedef __attribute__((ext_vector_type(8)))  __bf16 v8bf;
typedef __attribute__((ext_vector_type(8)))  float  v8f;

#define D_MODEL 1024
#define N_HEAD  16
#define HD      64
#define BATCH   4
#define SEQ     2048
#define MROWS   (BATCH * SEQ)   // 8192

__device__ __forceinline__ v8f wmma_bf16(v16bf a, v16bf b, v8f c) {
  return __builtin_amdgcn_wmma_f32_16x16x32_bf16(
      false, a, false, b, (short)0, c, false, false);
}

// ---- DPP 16-lane row reductions (stay within each 16-lane row of wave32) ---
template <int CTRL>
__device__ __forceinline__ float dpp_mov(float x) {
  return __int_as_float(__builtin_amdgcn_update_dpp(
      0, __float_as_int(x), CTRL, 0xf, 0xf, true));
}
__device__ __forceinline__ float rowmax16(float x) {
  x = fmaxf(x, dpp_mov<0x121>(x));   // row_ror:1
  x = fmaxf(x, dpp_mov<0x122>(x));   // row_ror:2
  x = fmaxf(x, dpp_mov<0x124>(x));   // row_ror:4
  x = fmaxf(x, dpp_mov<0x128>(x));   // row_ror:8
  return x;
}
__device__ __forceinline__ float rowsum16(float x) {
  x += dpp_mov<0x121>(x);
  x += dpp_mov<0x122>(x);
  x += dpp_mov<0x124>(x);
  x += dpp_mov<0x128>(x);
  return x;
}

// ---- async global -> LDS staging (16 bytes per lane), ASYNCcnt tracked ----
__device__ __forceinline__ void async_copy_b128(uint32_t lds_byte_addr,
                                                const void* gptr) {
  uint64_t ga = (uint64_t)(uintptr_t)gptr;
  asm volatile("global_load_async_to_lds_b128 %0, %1, off"
               :: "v"(lds_byte_addr), "v"(ga) : "memory");
}
__device__ __forceinline__ void async_wait0() {
  asm volatile("s_wait_asynccnt 0" ::: "memory");
}

// A-fragment (16x32, 16-bit): lane l<16 -> row l, K 0..7 / 16..23;
// lane l>=16 -> row l-16, K 8..15 / 24..31.  p -> (row0,k0), ld in elems.
__device__ __forceinline__ v16bf load_a16x32(const bf16* __restrict__ p, int ld) {
  const int lane = threadIdx.x & 31;
  const bf16* row = p + (size_t)(lane & 15) * ld + ((lane < 16) ? 0 : 8);
  v8bf lo = *(const v8bf*)(row);
  v8bf hi = *(const v8bf*)(row + 16);
  v16bf a;
#pragma unroll
  for (int i = 0; i < 8; ++i) { a[i] = lo[i]; a[i + 8] = hi[i]; }
  return a;
}

// B-fragment (32x16) from an LDS tile of BT rows (row = N index, cols = K).
// lane l<16 -> col l, K 0..7 / 16..23; lane l>=16 -> col l-16, K 8..15 / 24..31.
__device__ __forceinline__ v16bf lds_bT32x16(const char* tile, int row0,
                                             int rowstride_bytes, int kbyte0) {
  const int lane = threadIdx.x & 31;
  const char* p = tile + (size_t)(row0 + (lane & 15)) * rowstride_bytes +
                  kbyte0 + ((lane < 16) ? 0 : 16);
  v8bf lo = *(const v8bf*)(p);
  v8bf hi = *(const v8bf*)(p + 32);
  v16bf b;
#pragma unroll
  for (int i = 0; i < 8; ++i) { b[i] = lo[i]; b[i + 8] = hi[i]; }
  return b;
}

// ---------------------------------------------------------------------------
// fp32 -> bf16 conversions
// ---------------------------------------------------------------------------
__global__ void mha_cvt_bf16(const float* __restrict__ x, bf16* __restrict__ y, int n) {
  for (int i = blockIdx.x * blockDim.x + threadIdx.x; i < n; i += gridDim.x * blockDim.x)
    y[i] = (bf16)x[i];
}

__global__ void mha_cvt_bf16_T(const float* __restrict__ w, bf16* __restrict__ wt,
                               int rows, int cols) {
  const int n = rows * cols;
  for (int i = blockIdx.x * blockDim.x + threadIdx.x; i < n; i += gridDim.x * blockDim.x) {
    const int r = i / cols, c = i % cols;
    wt[(size_t)c * rows + r] = (bf16)w[i];
  }
}

// ---------------------------------------------------------------------------
// C = cscale * A(MxK bf16 row-major) x B, BT given as (NxK bf16).
// Block = 8 waves; wave w -> rows [bx*256+32w,+32) (2 A-tiles), cols
// [by*64,+64) (4 B-tiles) -> 8 WMMAs per 32-K chunk. Double-buffered
// 64-N x 64-K BT stages via async copies, shared by all waves.
// MODE: 1=[B,H,S,64] bf16, 2=[B,H,64,S] bf16, 3=fp32 row-major.
// ---------------------------------------------------------------------------
#define GT_PAD 144  // bytes per LDS row: 128 data + 16 pad

template <int MODE>
__global__ __launch_bounds__(256, 2)
void mha_gemm_wmma(const bf16* __restrict__ A, const bf16* __restrict__ BT,
                   void* __restrict__ C, int M, int N, int K, float cscale) {
  const int wave = threadIdx.x >> 5;
  const int lane = threadIdx.x & 31;
  const int tid = threadIdx.x;
  const int m0 = blockIdx.x * 256 + wave * 32;
  const int n0 = blockIdx.y * 64;

  __shared__ __align__(16) char btile[2 * 64 * GT_PAD];
  const uint32_t btile_lds = (uint32_t)(uintptr_t)btile;

  // one 64x64 stage = 64 rows x 128B = 8KB = 512 x 16B units (2 per thread)
  auto stage_b = [&](int buf, int k0) {
#pragma unroll
    for (int u = 0; u < 2; ++u) {
      const int unit = tid + u * 256;
      const int row = unit >> 3, chk = unit & 7;
      async_copy_b128(btile_lds + (uint32_t)(buf * 64 + row) * GT_PAD + chk * 16,
                      (const char*)(BT + (size_t)(n0 + row) * K + k0) + chk * 16);
    }
  };

  v8f acc[2][4];
#pragma unroll
  for (int mi = 0; mi < 2; ++mi)
#pragma unroll
    for (int t = 0; t < 4; ++t)
#pragma unroll
      for (int i = 0; i < 8; ++i) acc[mi][t][i] = 0.0f;

  stage_b(0, 0);
  int sidx = 0;
  for (int k = 0; k < K; k += 64, ++sidx) {
    const int cur = sidx & 1;
    async_wait0();       // our own stage copies done
    __syncthreads();     // everyone's copies done; prev stage reads done
    if (k + 64 < K) stage_b(1 - cur, k + 64);  // prefetch overlaps compute

    const char* bt = btile + (size_t)cur * 64 * GT_PAD;
#pragma unroll
    for (int kk = 0; kk < 2; ++kk) {
      v16bf a0 = load_a16x32(A + (size_t)m0 * K + k + 32 * kk, K);
      v16bf a1 = load_a16x32(A + (size_t)(m0 + 16) * K + k + 32 * kk, K);
#pragma unroll
      for (int t = 0; t < 4; ++t) {
        v16bf b = lds_bT32x16(bt, 16 * t, GT_PAD, 64 * kk);
        acc[0][t] = wmma_bf16(a0, b, acc[0][t]);
        acc[1][t] = wmma_bf16(a1, b, acc[1][t]);
      }
    }
  }

  const int col = lane & 15;
  const int rowbase = (lane < 16) ? 0 : 8;
#pragma unroll
  for (int mi = 0; mi < 2; ++mi) {
#pragma unroll
    for (int t = 0; t < 4; ++t) {
#pragma unroll
      for (int r = 0; r < 8; ++r) {
        const int gm = m0 + 16 * mi + rowbase + r;
        const int gn = n0 + 16 * t + col;
        const float val = acc[mi][t][r] * cscale;
        if (MODE == 1) {
          const int bb = gm >> 11, s = gm & (SEQ - 1);
          const int h = gn >> 6, d = gn & (HD - 1);
          ((bf16*)C)[(((size_t)(bb * N_HEAD + h)) * SEQ + s) * HD + d] = (bf16)val;
        } else if (MODE == 2) {
          const int bb = gm >> 11, s = gm & (SEQ - 1);
          const int h = gn >> 6, d = gn & (HD - 1);
          ((bf16*)C)[(((size_t)(bb * N_HEAD + h)) * HD + d) * SEQ + s] = (bf16)val;
        } else {
          ((float*)C)[(size_t)gm * N + gn] = val;
        }
      }
    }
  }
}

// ---------------------------------------------------------------------------
// Flash attention. Block = 8 waves sharing one (b,h); wave w owns the 16-query
// tile q0 = (bx*8+w)*16. Double-buffered 64-key stages: K tile (64 keys x 64d)
// and Vt tile (64d x 64 keys) staged via async copies, processed as two
// 32-key chunks. Q already carries 1/sqrt(d). ctx out: bf16 [B, S, H*64].
// ---------------------------------------------------------------------------
#define AT_PAD 144              // 128B row + 16B pad (both K and Vt tiles)
#define P_STRIDE 40             // P tile row stride in bf16 elems (32 + 8 pad)

__global__ __launch_bounds__(256, 2)
void mha_attn_wmma(const bf16* __restrict__ Qh, const bf16* __restrict__ Kh,
                   const bf16* __restrict__ Vt, bf16* __restrict__ ctx) {
  const int wave = threadIdx.x >> 5;
  const int lane = threadIdx.x & 31;
  const int tid = threadIdx.x;
  const int h = blockIdx.y;
  const int b = blockIdx.z;
  const int q0 = (blockIdx.x * 8 + wave) * 16;

  const bf16* Qb = Qh + ((size_t)(b * N_HEAD + h)) * SEQ * HD;
  const bf16* Kb = Kh + ((size_t)(b * N_HEAD + h)) * SEQ * HD;
  const bf16* Vb = Vt + ((size_t)(b * N_HEAD + h)) * HD * SEQ;

  // [2 x Ktile][2 x Vtile][8 x P-tile]
  __shared__ __align__(16) char smem[4 * 64 * AT_PAD + 8 * 16 * P_STRIDE * 2];
  char* kbuf0 = smem;
  char* vbuf0 = smem + 2 * 64 * AT_PAD;
  bf16* pw = (bf16*)(smem + 4 * 64 * AT_PAD) + (size_t)wave * 16 * P_STRIDE;
  const uint32_t kbuf_lds = (uint32_t)(uintptr_t)kbuf0;
  const uint32_t vbuf_lds = (uint32_t)(uintptr_t)vbuf0;

  // one stage: K tile 64 rows x 128B + Vt tile 64 rows x 128B (2+2 units/thr)
  auto stage_kv = [&](int buf, int kb0) {
#pragma unroll
    for (int u = 0; u < 2; ++u) {
      const int unit = tid + u * 256;
      const int row = unit >> 3, chk = unit & 7;
      async_copy_b128(kbuf_lds + (uint32_t)(buf * 64 + row) * AT_PAD + chk * 16,
                      (const char*)(Kb + (size_t)(kb0 + row) * HD) + chk * 16);
    }
#pragma unroll
    for (int u = 0; u < 2; ++u) {
      const int unit = tid + u * 256;
      const int row = unit >> 3, chk = unit & 7;
      async_copy_b128(vbuf_lds + (uint32_t)(buf * 64 + row) * AT_PAD + chk * 16,
                      (const char*)(Vb + (size_t)row * SEQ + kb0) + chk * 16);
    }
  };

  const v16bf qa0 = load_a16x32(Qb + (size_t)q0 * HD + 0, HD);
  const v16bf qa1 = load_a16x32(Qb + (size_t)q0 * HD + 32, HD);

  v8f acc[4];
#pragma unroll
  for (int t = 0; t < 4; ++t)
#pragma unroll
    for (int i = 0; i < 8; ++i) acc[t][i] = 0.0f;

  float mrow[8], lrow[8];
#pragma unroll
  for (int r = 0; r < 8; ++r) { mrow[r] = -1e30f; lrow[r] = 0.0f; }

  const int col = lane & 15;
  const int rowbase = (lane < 16) ? 0 : 8;

  stage_kv(0, 0);
  int sidx = 0;
  for (int kb = 0; kb < SEQ; kb += 64, ++sidx) {
    const int cur = sidx & 1;
    async_wait0();
    __syncthreads();
    if (kb + 64 < SEQ) stage_kv(1 - cur, kb + 64);  // overlap with compute

    const char* kt = kbuf0 + (size_t)cur * 64 * AT_PAD;
    const char* vt = vbuf0 + (size_t)cur * 64 * AT_PAD;

#pragma unroll
    for (int c = 0; c < 2; ++c) {   // two 32-key chunks per stage
      // ---- scores: S1 = keys 32c..32c+15, S2 = keys 32c+16..32c+31
      v8f s1, s2;
#pragma unroll
      for (int i = 0; i < 8; ++i) { s1[i] = 0.0f; s2[i] = 0.0f; }
      {
        v16bf k00 = lds_bT32x16(kt, 32 * c + 0,  AT_PAD, 0);
        v16bf k01 = lds_bT32x16(kt, 32 * c + 0,  AT_PAD, 64);
        s1 = wmma_bf16(qa0, k00, s1);
        s1 = wmma_bf16(qa1, k01, s1);
        v16bf k10 = lds_bT32x16(kt, 32 * c + 16, AT_PAD, 0);
        v16bf k11 = lds_bT32x16(kt, 32 * c + 16, AT_PAD, 64);
        s2 = wmma_bf16(qa0, k10, s2);
        s2 = wmma_bf16(qa1, k11, s2);
      }

      // ---- online softmax (DPP row reductions; Q pre-scaled by 1/sqrt(d))
      float mloc[8];
#pragma unroll
      for (int r = 0; r < 8; ++r)
        mloc[r] = rowmax16(fmaxf(s1[r], s2[r]));
      float corr[8], p1[8], p2[8];
#pragma unroll
      for (int r = 0; r < 8; ++r) {
        const float mn = fmaxf(mrow[r], mloc[r]);
        corr[r] = __expf(mrow[r] - mn);
        mrow[r] = mn;
        p1[r] = __expf(s1[r] - mn);
        p2[r] = __expf(s2[r] - mn);
        lrow[r] = lrow[r] * corr[r] + rowsum16(p1[r] + p2[r]);
      }
#pragma unroll
      for (int t = 0; t < 4; ++t)
#pragma unroll
        for (int r = 0; r < 8; ++r) acc[t][r] *= corr[r];

      // ---- re-fragment P (16x32) through wave-private LDS tile
#pragma unroll
      for (int r = 0; r < 8; ++r) {
        pw[(rowbase + r) * P_STRIDE + col]      = (bf16)p1[r];
        pw[(rowbase + r) * P_STRIDE + 16 + col] = (bf16)p2[r];
      }
      v16bf pa;
      {
        const bf16* prow = pw + (size_t)(lane & 15) * P_STRIDE +
                           ((lane < 16) ? 0 : 8);
        v8bf lo = *(const v8bf*)(prow);
        v8bf hi = *(const v8bf*)(prow + 16);
#pragma unroll
        for (int i = 0; i < 8; ++i) { pa[i] = lo[i]; pa[i + 8] = hi[i]; }
      }

      // ---- ctx += P(16x32) x V(32x64); Vt tile rows = d, cols = 64 keys
#pragma unroll
      for (int t = 0; t < 4; ++t) {
        v16bf vb = lds_bT32x16(vt, 16 * t, AT_PAD, 64 * c);
        acc[t] = wmma_bf16(pa, vb, acc[t]);
      }
    }
  }

  // ---- epilogue
#pragma unroll
  for (int r = 0; r < 8; ++r) lrow[r] = 1.0f / lrow[r];
#pragma unroll
  for (int t = 0; t < 4; ++t)
#pragma unroll
    for (int r = 0; r < 8; ++r) {
      const int qrow = q0 + rowbase + r;
      const int d = 16 * t + col;
      ctx[((size_t)(b * SEQ) + qrow) * D_MODEL + h * HD + d] =
          (bf16)(acc[t][r] * lrow[r]);
    }
}

// ---------------------------------------------------------------------------
extern "C" void kernel_launch(void* const* d_in, const int* in_sizes, int n_in,
                              void* d_out, int out_size, void* d_ws, size_t ws_size,
                              hipStream_t stream) {
  (void)in_sizes; (void)n_in; (void)out_size; (void)ws_size;
  const float* q  = (const float*)d_in[0];
  const float* k  = (const float*)d_in[1];
  const float* v  = (const float*)d_in[2];
  const float* Wq = (const float*)d_in[3];
  const float* Wk = (const float*)d_in[4];
  const float* Wv = (const float*)d_in[5];
  const float* Wo = (const float*)d_in[6];

  char* ws = (char*)d_ws;
  size_t off = 0;
  auto take = [&](size_t bytes) -> char* {
    char* p = ws + off;
    off += (bytes + 255) & ~(size_t)255;
    return p;
  };

  const size_t MK = (size_t)MROWS * D_MODEL;
  const size_t WW = (size_t)D_MODEL * D_MODEL;
  bf16* qx  = (bf16*)take(MK * sizeof(bf16));
  bf16* kx  = (bf16*)take(MK * sizeof(bf16));
  bf16* vx  = (bf16*)take(MK * sizeof(bf16));
  bf16* WqT = (bf16*)take(WW * sizeof(bf16));
  bf16* WkT = (bf16*)take(WW * sizeof(bf16));
  bf16* WvT = (bf16*)take(WW * sizeof(bf16));
  bf16* WoT = (bf16*)take(WW * sizeof(bf16));
  bf16* Qh  = (bf16*)take(MK * sizeof(bf16));   // [B,H,S,64] (pre-scaled)
  bf16* Kh  = (bf16*)take(MK * sizeof(bf16));   // [B,H,S,64]
  bf16* Vt  = (bf16*)take(MK * sizeof(bf16));   // [B,H,64,S]
  bf16* ctx = qx;  // qx dead after Q projection -> reuse as ctx [B,S,1024]

  const dim3 cb(256);
  mha_cvt_bf16<<<4096, cb, 0, stream>>>(q, qx, (int)MK);
  mha_cvt_bf16<<<4096, cb, 0, stream>>>(k, kx, (int)MK);
  mha_cvt_bf16<<<4096, cb, 0, stream>>>(v, vx, (int)MK);
  mha_cvt_bf16_T<<<2048, cb, 0, stream>>>(Wq, WqT, D_MODEL, D_MODEL);
  mha_cvt_bf16_T<<<2048, cb, 0, stream>>>(Wk, WkT, D_MODEL, D_MODEL);
  mha_cvt_bf16_T<<<2048, cb, 0, stream>>>(Wv, WvT, D_MODEL, D_MODEL);
  mha_cvt_bf16_T<<<2048, cb, 0, stream>>>(Wo, WoT, D_MODEL, D_MODEL);

  const dim3 gg(MROWS / 256, D_MODEL / 64);
  mha_gemm_wmma<1><<<gg, cb, 0, stream>>>(qx, WqT, Qh, MROWS, D_MODEL, D_MODEL, 0.125f);
  mha_gemm_wmma<1><<<gg, cb, 0, stream>>>(kx, WkT, Kh, MROWS, D_MODEL, D_MODEL, 1.0f);
  mha_gemm_wmma<2><<<gg, cb, 0, stream>>>(vx, WvT, Vt, MROWS, D_MODEL, D_MODEL, 1.0f);

  const dim3 ga(SEQ / 16 / 8, N_HEAD, BATCH);
  mha_attn_wmma<<<ga, cb, 0, stream>>>(Qh, Kh, Vt, ctx);

  mha_gemm_wmma<3><<<gg, cb, 0, stream>>>(ctx, WoT, d_out, MROWS, D_MODEL, D_MODEL, 1.0f);
}
